// CNN_LSTM_76596446757029
// MI455X (gfx1250) — matrix operand order repaired and checked by reading, not compiled, not run on hardware
//
#include <hip/hip_runtime.h>

// ---------------------------------------------------------------------------
// CDNA5 (gfx1250) LSTM sequence predictor.
// B=256 batch, H=2048 hidden, 80 observed + 80 autoregressive steps.
// Per step: one fused kernel computes gates = h_bf16 @ Whh_bf16^T via
// v_wmma_f32_16x16x32_bf16 with i/f/g/o for each (row, unit) tile held in
// the WMMA accumulators, applies the LSTM cell in registers, and writes
// c (f32) + h (bf16, double-buffered).  A second small kernel reduces
// h . W_lin into the prediction (deterministic tree reduction).
// ---------------------------------------------------------------------------

typedef __attribute__((ext_vector_type(16))) __bf16 v16bf;
typedef __attribute__((ext_vector_type(8)))  float  v8f;

#define HH    2048      // hidden size
#define G4    8192      // 4*H gate width
#define NBAT  256       // batch
#define TOBS  80        // observed steps
#define TTOT  160       // observed + future

union FragU {
    uint4  q[2];
    v16bf  v;
};

// A-matrix 16x32 bf16 fragment (ISA 7.12.2): lane&15 = row M,
// lanes 0-15 hold K {0..7, 16..23}, lanes 16-31 hold K {8..15, 24..31}.
// Caller passes kb = k0 + (lane>>4)*8.  Source row-major [*, HH] bf16.
__device__ __forceinline__ v16bf load_fragA(const __bf16* __restrict__ p,
                                            int row, int kb) {
    FragU f;
    const char* base = (const char*)(p + (size_t)row * HH + kb);
    f.q[0] = *(const uint4*)(base);        // K = kb .. kb+7
    f.q[1] = *(const uint4*)(base + 32);   // K = kb+16 .. kb+23
    return f.v;
}

// B-matrix 32x16 bf16 fragment: lane&15 = column N (a row of Whh, since
// B = Whh^T), lanes 0-15 hold K {0..7, 16..23}, lanes 16-31 hold K {8..15, 24..31}
// (same order as A).  Caller passes kb = k0 + (lane>>4)*8.
__device__ __forceinline__ v16bf load_fragB(const __bf16* __restrict__ p,
                                            int row, int kb) {
    FragU f;
    const char* base = (const char*)(p + (size_t)row * HH + kb);
    f.q[0] = *(const uint4*)(base);        // K = kb .. kb+7
    f.q[1] = *(const uint4*)(base + 32);   // K = kb+16 .. kb+23
    return f.v;
}

__device__ __forceinline__ float sigmoidf_fast(float v) {
    return 1.0f / (1.0f + __expf(-v));
}

// ---------------------------------------------------------------------------
// Fused LSTM step.  Wave tile: 32 batch rows x 16 hidden units x 4 gates
// (= 8 WMMA accumulators).  Block = 4 waves -> 32 rows x 64 units.
// Grid: (256/32) x (2048/64) = 8 x 32.
// The four gate tiles for a unit block live at Whh rows {u, H+u, 2H+u, 3H+u};
// after the K loop the accumulators ARE i/f/g/o, so the cell update happens
// in registers and no gates buffer ever touches memory.
// h is double-buffered (read hA from step t-1, write hOut for step t).
// ---------------------------------------------------------------------------
__global__ __launch_bounds__(128)
void lstm_step_fused(const __bf16* __restrict__ hA,    // [256, 2048] step t-1
                     __bf16* __restrict__ hOut,        // [256, 2048] step t
                     const __bf16* __restrict__ W,     // [8192, 2048] bf16
                     float* __restrict__ c,            // [256, 2048]
                     const float* __restrict__ x,      // [256, 80]
                     const float* __restrict__ Wih,    // [8192]
                     const float* __restrict__ bias,   // [8192]
                     const float* __restrict__ out,    // [256, 160] (read t-1)
                     int t) {
    const int lane = threadIdx.x & 31;
    const int wave = threadIdx.x >> 5;
    const int l15  = lane & 15;
    const int half = lane >> 4;

    const int m0 = blockIdx.x * 32;                // batch row block
    const int n0 = blockIdx.y * 64 + wave * 16;    // hidden unit block

    // Stage the 32 per-row scalar inputs for this block through LDS.
    __shared__ float xs[32];
    if (threadIdx.x < 32) {
        const int row = m0 + threadIdx.x;
        xs[threadIdx.x] = (t < TOBS) ? x[(size_t)row * TOBS + t]
                                     : out[(size_t)row * TTOT + (t - 1)];
    }
    __syncthreads();

    v8f acc[2][4] = {};   // [m-subtile][gate i,f,g,o]

    const int kbA = half * 8;
    const int kbB = half * 8;
    const int ar0 = m0 + l15;
    const int ar1 = m0 + 16 + l15;
    const int br  = n0 + l15;

    for (int k0 = 0; k0 < HH; k0 += 32) {
        v16bf a0 = load_fragA(hA, ar0, k0 + kbA);
        v16bf a1 = load_fragA(hA, ar1, k0 + kbA);
        v16bf bi = load_fragB(W, br,            k0 + kbB);
        v16bf bf = load_fragB(W, HH     + br,   k0 + kbB);
        v16bf bg = load_fragB(W, 2 * HH + br,   k0 + kbB);
        v16bf bo = load_fragB(W, 3 * HH + br,   k0 + kbB);

        acc[0][0] = __builtin_amdgcn_wmma_f32_16x16x32_bf16(false, a0, false, bi, (short)0, acc[0][0], false, false);
        acc[0][1] = __builtin_amdgcn_wmma_f32_16x16x32_bf16(false, a0, false, bf, (short)0, acc[0][1], false, false);
        acc[0][2] = __builtin_amdgcn_wmma_f32_16x16x32_bf16(false, a0, false, bg, (short)0, acc[0][2], false, false);
        acc[0][3] = __builtin_amdgcn_wmma_f32_16x16x32_bf16(false, a0, false, bo, (short)0, acc[0][3], false, false);
        acc[1][0] = __builtin_amdgcn_wmma_f32_16x16x32_bf16(false, a1, false, bi, (short)0, acc[1][0], false, false);
        acc[1][1] = __builtin_amdgcn_wmma_f32_16x16x32_bf16(false, a1, false, bf, (short)0, acc[1][1], false, false);
        acc[1][2] = __builtin_amdgcn_wmma_f32_16x16x32_bf16(false, a1, false, bg, (short)0, acc[1][2], false, false);
        acc[1][3] = __builtin_amdgcn_wmma_f32_16x16x32_bf16(false, a1, false, bo, (short)0, acc[1][3], false, false);
    }

    // Per-unit gate parameters (lane&15 selects the unit column).
    const int unit = n0 + l15;
    float wih_g[4], bia_g[4];
#pragma unroll
    for (int g = 0; g < 4; ++g) {
        wih_g[g] = Wih[g * HH + unit];
        bia_g[g] = bias[g * HH + unit];
    }

    // C/D layout: VGPR v -> row m0 + mi*16 + 8*half + v, lane&15 -> column.
#pragma unroll
    for (int mi = 0; mi < 2; ++mi) {
#pragma unroll
        for (int v = 0; v < 8; ++v) {
            const int rloc = mi * 16 + half * 8 + v;
            const int row  = m0 + rloc;
            const float xv = xs[rloc];
            const size_t idx = (size_t)row * HH + unit;

            float gi = acc[mi][0][v] + xv * wih_g[0] + bia_g[0];
            float gf = acc[mi][1][v] + xv * wih_g[1] + bia_g[1];
            float gg = acc[mi][2][v] + xv * wih_g[2] + bia_g[2];
            float go = acc[mi][3][v] + xv * wih_g[3] + bia_g[3];

            float cn = sigmoidf_fast(gf) * c[idx] + sigmoidf_fast(gi) * tanhf(gg);
            c[idx]   = cn;
            hOut[idx] = (__bf16)(sigmoidf_fast(go) * tanhf(cn));
        }
    }
}

// ---------------------------------------------------------------------------
// Projection: out[b, t] = h[b, :] . W_lin + b_lin.  One block per batch row,
// deterministic LDS tree reduction (no float atomics -> bitwise reproducible
// across graph replays).
// ---------------------------------------------------------------------------
__global__ __launch_bounds__(256)
void lstm_proj(const __bf16* __restrict__ hb,
               const float* __restrict__ Wlin,
               const float* __restrict__ blin,
               float* __restrict__ out,
               int t) {
    const int b   = blockIdx.x;
    const int tid = threadIdx.x;
    const __bf16* hrow = hb + (size_t)b * HH;

    float partial = 0.0f;
    for (int u = tid; u < HH; u += 256)
        partial += (float)hrow[u] * Wlin[u];

    __shared__ float red[256];
    red[tid] = partial;
    __syncthreads();
#pragma unroll
    for (int s = 128; s > 0; s >>= 1) {
        if (tid < s) red[tid] += red[tid + s];
        __syncthreads();
    }
    if (tid == 0)
        out[(size_t)b * TTOT + t] = red[0] + blin[0];
}

// ---------------------------------------------------------------------------
// Prep: convert W_hh f32 -> bf16 (RNE via hardware cvt).
// ---------------------------------------------------------------------------
__global__ void prep_whh(const float* __restrict__ w, __bf16* __restrict__ wb) {
    const size_t n = (size_t)G4 * HH;
    const size_t stride = (size_t)gridDim.x * blockDim.x;
    for (size_t i = (size_t)blockIdx.x * blockDim.x + threadIdx.x; i < n; i += stride)
        wb[i] = (__bf16)w[i];
}

// Prep: bias = b_ih + b_hh; zero c and h buffer 0 (workspace is poisoned).
__global__ void prep_state(const float* __restrict__ bih,
                           const float* __restrict__ bhh,
                           float* __restrict__ bias,
                           float* __restrict__ c,
                           __bf16* __restrict__ hb0) {
    const int n = NBAT * HH;
    const int stride = gridDim.x * blockDim.x;
    for (int i = blockIdx.x * blockDim.x + threadIdx.x; i < n; i += stride) {
        c[i]   = 0.0f;
        hb0[i] = (__bf16)0.0f;
        if (i < G4) bias[i] = bih[i] + bhh[i];
    }
}

// ---------------------------------------------------------------------------
static inline size_t align256(size_t x) { return (x + 255) & ~(size_t)255; }

extern "C" void kernel_launch(void* const* d_in, const int* in_sizes, int n_in,
                              void* d_out, int out_size, void* d_ws, size_t ws_size,
                              hipStream_t stream) {
    (void)in_sizes; (void)n_in; (void)out_size; (void)ws_size;

    const float* x    = (const float*)d_in[0];   // [256, 80]
    const float* Wih  = (const float*)d_in[1];   // [8192, 1]
    const float* Whh  = (const float*)d_in[2];   // [8192, 2048]
    const float* bih  = (const float*)d_in[3];   // [8192]
    const float* bhh  = (const float*)d_in[4];   // [8192]
    const float* Wlin = (const float*)d_in[5];   // [1, 2048]
    const float* blin = (const float*)d_in[6];   // [1]
    float* out = (float*)d_out;                  // [256, 160]

    // Workspace carve-up (~37.6 MB total).
    char* ws = (char*)d_ws;
    size_t off = 0;
    __bf16* whh_bf = (__bf16*)(ws + off); off = align256(off + (size_t)G4 * HH * 2);
    float*  cstate = (float*) (ws + off); off = align256(off + (size_t)NBAT * HH * 4);
    __bf16* h_bufA = (__bf16*)(ws + off); off = align256(off + (size_t)NBAT * HH * 2);
    __bf16* h_bufB = (__bf16*)(ws + off); off = align256(off + (size_t)NBAT * HH * 2);
    float*  bias   = (float*) (ws + off); off = align256(off + (size_t)G4 * 4);

    // One-time prep (runs every launch; deterministic).
    prep_whh<<<4096, 256, 0, stream>>>(Whh, whh_bf);
    prep_state<<<2048, 256, 0, stream>>>(bih, bhh, bias, cstate, h_bufA);

    // 160 sequential LSTM steps.  h double-buffered: step t reads h[t&1],
    // writes h[(t+1)&1]; projection reads the freshly written buffer.
    __bf16* hbuf[2] = { h_bufA, h_bufB };
    const dim3 step_grid(NBAT / 32, HH / 64);
    for (int t = 0; t < TTOT; ++t) {
        __bf16* h_in  = hbuf[t & 1];
        __bf16* h_out = hbuf[(t + 1) & 1];
        lstm_step_fused<<<step_grid, 128, 0, stream>>>(h_in, h_out, whh_bf,
                                                       cstate, x, Wih, bias,
                                                       out, t);
        lstm_proj<<<NBAT, 256, 0, stream>>>(h_out, Wlin, blin, out, t);
    }
}
